// MultiHeadEFRALayer_NoMP_2396591751590
// MI455X (gfx1250) — matrix operand order repaired and checked by reading, not compiled, not run on hardware
//
#include <hip/hip_runtime.h>

#define N_GENE  16384
#define N_TF    512
#define D_MODEL 512
#define HD      512   // N_HEADS * D_K
#define N_HEADS 8
#define D_K     64

// ---- output layout (flat float offsets) ----
#define OFF_HTF 0
#define CNT_HTF (N_TF * D_MODEL)                 // 262144
#define OFF_HG  (OFF_HTF + CNT_HTF)              // 262144
#define CNT_HG  (N_GENE * D_MODEL)               // 8388608
#define OFF_A   (OFF_HG + CNT_HG)                // 8650752
#define CNT_A   (N_TF * N_GENE)                  // 8388608
#define OFF_U   (OFF_A + CNT_A)                  // 17039360
#define CNT_U   (N_TF * N_GENE)
#define OFF_AL  (OFF_U + CNT_U)                  // 25427968
#define CNT_AL  (N_TF * N_GENE * 3)              // 25165824

typedef __attribute__((ext_vector_type(16))) __bf16 v16bf;
typedef __attribute__((ext_vector_type(8)))  float  v8f;

// float -> bf16, truncation (1 VALU op)
__device__ __forceinline__ __bf16 f2bf(float x) {
    union { float f; unsigned u; } v; v.f = x;
    union { unsigned short s; __bf16 b; } o; o.s = (unsigned short)(v.u >> 16);
    return o.b;
}

// two floats -> packed {bf16(hi), bf16(lo)} in one u32
__device__ __forceinline__ unsigned pack2bf(float lo, float hi) {
    union { float f; unsigned u; } a, b; a.f = lo; b.f = hi;
    return (a.u >> 16) | (b.u & 0xffff0000u);
}

// A fragment (16x32 bf16, M x K) from LDS tile, row-major, leading dim `ld`.
// lane 0-15: M=lane,   K = kbase+{0..7} U kbase+{16..23}
// lane16-31: M=lane-16,K = kbase+{8..15} U kbase+{24..31}
__device__ __forceinline__ v16bf load_a_lds(const __bf16* s, int ld, int kbase, int lane) {
    int m  = lane & 15;
    int k0 = kbase + ((lane >> 4) << 3);
    const __bf16* p = s + m * ld + k0;
    v16bf a;
#pragma unroll
    for (int i = 0; i < 8; ++i) { a[i] = p[i]; a[8 + i] = p[16 + i]; }
    return a;
}

// B fragment (32x16 bf16, K x N); N-major matrix stored row-major [n][k],
// leading dim ld -> two contiguous 16-byte loads per lane.
__device__ __forceinline__ v16bf load_b_rowmajor(const __bf16* mat, int ld,
                                                 int nbase, int kbase, int lane) {
    int n  = nbase + (lane & 15);
    int k0 = kbase + ((lane >> 4) << 3);
    const __bf16* p = mat + (size_t)n * ld + k0;
    v16bf b;
#pragma unroll
    for (int i = 0; i < 8; ++i) { b[i] = p[i]; b[8 + i] = p[16 + i]; }
    return b;
}

// ---------------------------------------------------------------------------
// One-time: Wt[n][k] = bf16(W[k][n]); LDS 32x32 tile transpose, coalesced.
// ---------------------------------------------------------------------------
__global__ __launch_bounds__(256)
void wtrans_kernel(const float* __restrict__ W, __bf16* __restrict__ Wt) {
    __shared__ __bf16 tile[32][33];
    const int tx = threadIdx.x & 31;
    const int ty = threadIdx.x >> 5;          // 0..7
    const int kb = blockIdx.y * 32, nb = blockIdx.x * 32;
#pragma unroll
    for (int j = 0; j < 32; j += 8)
        tile[ty + j][tx] = f2bf(W[(size_t)(kb + ty + j) * HD + nb + tx]);
    __syncthreads();
#pragma unroll
    for (int j = 0; j < 32; j += 8)
        Wt[(size_t)(nb + ty + j) * D_MODEL + kb + tx] = tile[tx][ty + j];
}

// ---------------------------------------------------------------------------
// P[M x 512] (bf16) = X[M x 512] (f32, optionally gathered by idx) @ Wt^T
// block = 256 threads = 8 waves; tile = 16 (M) x 128 (N); 16 WMMA k-steps.
// ---------------------------------------------------------------------------
__global__ __launch_bounds__(256)
void proj_gemm_kernel(const float* __restrict__ X, const __bf16* __restrict__ Wt,
                      const int* __restrict__ idx, __bf16* __restrict__ P) {
    const int tid   = threadIdx.x;
    const int lane  = tid & 31;
    const int wave  = tid >> 5;
    const int mbase = blockIdx.y * 16;
    const int nbase = blockIdx.x * 128 + wave * 16;

    __shared__ __bf16 sA[16 * D_MODEL];       // 16 KB
    unsigned* sA32 = (unsigned*)sA;
#pragma unroll
    for (int it = 0; it < 8; ++it) {
        int i  = tid + it * 256;              // float4 index in tile
        int m  = i >> 7;                      // 128 float4 per row
        int k4 = i & 127;
        int row = mbase + m;
        int src = idx ? idx[row] : row;
        float4 x = ((const float4*)(X + (size_t)src * D_MODEL))[k4];
        sA32[m * 256 + k4 * 2]     = pack2bf(x.x, x.y);
        sA32[m * 256 + k4 * 2 + 1] = pack2bf(x.z, x.w);
    }
    __syncthreads();

    v8f c = {};
#pragma unroll 4
    for (int ks = 0; ks < D_MODEL; ks += 32) {
        v16bf a = load_a_lds(sA, D_MODEL, ks, lane);
        v16bf b = load_b_rowmajor(Wt, D_MODEL, nbase, ks, lane);
        c = __builtin_amdgcn_wmma_f32_16x16x32_bf16(false, a, false, b,
                                                    (short)0, c, false, false);
    }

    const int n  = lane & 15;
    const int m0 = (lane >> 4) * 8;
#pragma unroll
    for (int r = 0; r < 8; ++r)
        P[(size_t)(mbase + m0 + r) * HD + nbase + n] = f2bf(c[r]);
}

// ---------------------------------------------------------------------------
// ALL heads fused: per (t,g) tile loop h=0..7, 3 streams x 2 WMMA each,
// 3x3 gate + softmax3; u[h] -> ws, head-mean u/alpha kept in registers and
// stored ONCE (pure stores, no RMW, no memset needed).
// block = 256 threads = 8 waves; tile = 16 (t) x 128 (g); 48 WMMAs/wave.
// ---------------------------------------------------------------------------
__global__ __launch_bounds__(256)
void score_gate_kernel(const __bf16* __restrict__ Qbf, const __bf16* __restrict__ Kbf,
                       const float* __restrict__ gw, const float* __restrict__ gb,
                       float* __restrict__ u_ws,
                       float* __restrict__ outU, float* __restrict__ outAl) {
    const int tid   = threadIdx.x;
    const int lane  = tid & 31;
    const int wave  = tid >> 5;
    const int tbase = blockIdx.y * 16;
    const int gtile = blockIdx.x * 128 + wave * 16;

    // stage Q rows (all heads, 3 streams): 3*16*512 bf16 = 48 KB, uint4 copies
    __shared__ __bf16 sQ[3][16 * HD];
    {
        uint4* d4 = (uint4*)&sQ[0][0];
        for (int i = tid; i < 3 * 16 * (HD / 8); i += 256) {   // 3072 uint4
            int e = i >> 10, rem = i & 1023;
            int m = rem >> 6, q = rem & 63;
            d4[i] = ((const uint4*)(Qbf + ((size_t)e * N_TF + tbase + m) * HD))[q];
        }
    }
    // prefetch this wave's K rows while staging Q
    {
        const __bf16* kp = Kbf + (size_t)(gtile + (lane & 15)) * HD;
        __builtin_prefetch(kp, 0, 3);
        __builtin_prefetch(kp + (size_t)N_GENE * HD, 0, 3);
        __builtin_prefetch(kp + (size_t)2 * N_GENE * HD, 0, 3);
    }
    __syncthreads();

    const int n  = lane & 15;
    const int m0 = (lane >> 4) * 8;
    const int gg = gtile + n;

    float uS[8]  = {0, 0, 0, 0, 0, 0, 0, 0};
    float a0S[8] = {0, 0, 0, 0, 0, 0, 0, 0};
    float a1S[8] = {0, 0, 0, 0, 0, 0, 0, 0};
    float a2S[8] = {0, 0, 0, 0, 0, 0, 0, 0};

    for (int h = 0; h < N_HEADS; ++h) {
        const int hoff = h * D_K;

        float G[3][3], B3[3];
#pragma unroll
        for (int e = 0; e < 3; ++e)
#pragma unroll
            for (int f = 0; f < 3; ++f) G[e][f] = gw[h * 9 + e * 3 + f];
#pragma unroll
        for (int f = 0; f < 3; ++f) B3[f] = gb[h * 3 + f];

        v8f acc[3];
#pragma unroll
        for (int e = 0; e < 3; ++e) {
            v8f c = {};
            const __bf16* Ke = Kbf + (size_t)e * N_GENE * HD;
#pragma unroll
            for (int ks = 0; ks < D_K; ks += 32) {
                v16bf a = load_a_lds(&sQ[e][0], HD, hoff + ks, lane);
                v16bf b = load_b_rowmajor(Ke, HD, gtile, hoff + ks, lane);
                c = __builtin_amdgcn_wmma_f32_16x16x32_bf16(false, a, false, b,
                                                            (short)0, c, false, false);
            }
            acc[e] = c;
        }

        float* uh = u_ws + (size_t)h * N_TF * N_GENE;
#pragma unroll
        for (int r = 0; r < 8; ++r) {
            const float s0 = acc[0][r] * 0.125f;   // 1/sqrt(D_K)
            const float s1 = acc[1][r] * 0.125f;
            const float s2 = acc[2][r] * 0.125f;

            float l0 = s0 * G[0][0] + s1 * G[1][0] + s2 * G[2][0] + B3[0];
            float l1 = s0 * G[0][1] + s1 * G[1][1] + s2 * G[2][1] + B3[1];
            float l2 = s0 * G[0][2] + s1 * G[1][2] + s2 * G[2][2] + B3[2];
            float mx = fmaxf(l0, fmaxf(l1, l2));
            float e0 = __expf(l0 - mx), e1 = __expf(l1 - mx), e2 = __expf(l2 - mx);
            float inv = 1.0f / (e0 + e1 + e2);
            float a0 = e0 * inv, a1 = e1 * inv, a2 = e2 * inv;
            float u  = a0 * s0 + a1 * s1 + a2 * s2;

            uh[(size_t)(tbase + m0 + r) * N_GENE + gg] = u;
            uS[r]  += u;
            a0S[r] += a0; a1S[r] += a1; a2S[r] += a2;
        }
    }

    // head means: single pure store per element
#pragma unroll
    for (int r = 0; r < 8; ++r) {
        const size_t ix = (size_t)(tbase + m0 + r) * N_GENE + gg;
        outU[ix] = uS[r] * 0.125f;
        outAl[ix * 3 + 0] = a0S[r] * 0.125f;
        outAl[ix * 3 + 1] = a1S[r] * 0.125f;
        outAl[ix * 3 + 2] = a2S[r] * 0.125f;
    }
}

// ---------------------------------------------------------------------------
// Per (h,t) row: max and 1/sum(exp) over 16384 genes -> stats[row] = {mx, inv}
// ---------------------------------------------------------------------------
__global__ __launch_bounds__(256)
void row_stats_kernel(const float* __restrict__ u, float* __restrict__ stats) {
    const int row = blockIdx.x;                 // h * N_TF + t
    const int tid = threadIdx.x;
    const float* p = u + (size_t)row * N_GENE;
    __shared__ float red[256];

    float mx = -3.4e38f;
    for (int g = tid; g < N_GENE; g += 256) mx = fmaxf(mx, p[g]);
    red[tid] = mx; __syncthreads();
    for (int s = 128; s > 0; s >>= 1) {
        if (tid < s) red[tid] = fmaxf(red[tid], red[tid + s]);
        __syncthreads();
    }
    mx = red[0]; __syncthreads();

    float sum = 0.f;
    for (int g = tid; g < N_GENE; g += 256) sum += __expf(p[g] - mx);
    red[tid] = sum; __syncthreads();
    for (int s = 128; s > 0; s >>= 1) {
        if (tid < s) red[tid] += red[tid + s];
        __syncthreads();
    }
    if (tid == 0) { stats[row * 2] = mx; stats[row * 2 + 1] = 1.0f / red[0]; }
}

// ---------------------------------------------------------------------------
// A_mean[t][g] = (1/8) sum_h exp(u[h][t][g]-mx[h][t]) * inv[h][t] -- one store
// ---------------------------------------------------------------------------
__global__ __launch_bounds__(256)
void amean_kernel(const float* __restrict__ u, const float* __restrict__ stats,
                  float* __restrict__ outA) {
    const int t = blockIdx.y;
    const int g = blockIdx.x * 256 + threadIdx.x;
    float acc = 0.f;
#pragma unroll
    for (int h = 0; h < N_HEADS; ++h) {
        const float mx  = stats[(h * N_TF + t) * 2];        // uniform -> scalar
        const float inv = stats[(h * N_TF + t) * 2 + 1];
        acc += __expf(u[((size_t)h * N_TF + t) * N_GENE + g] - mx) * inv;
    }
    outA[(size_t)t * N_GENE + g] = acc * 0.125f;
}

// ---------------------------------------------------------------------------
extern "C" void kernel_launch(void* const* d_in, const int* in_sizes, int n_in,
                              void* d_out, int out_size, void* d_ws, size_t ws_size,
                              hipStream_t stream) {
    const float* H_TF = (const float*)d_in[0];
    const float* H_G  = (const float*)d_in[1];
    // evidence order matches reference: 0=seq(bind), 1=exp(coexpr), 2=txt(know)
    const float* z[3]  = { (const float*)d_in[3], (const float*)d_in[2], (const float*)d_in[4] };
    const int*  tf_idx = (const int*)d_in[5];
    const float* Wq[3] = { (const float*)d_in[6], (const float*)d_in[8],  (const float*)d_in[10] };
    const float* Wk[3] = { (const float*)d_in[7], (const float*)d_in[9],  (const float*)d_in[11] };
    const float* gw = (const float*)d_in[12];
    const float* gb = (const float*)d_in[13];
    float* out = (float*)d_out;

    // workspace: Kbf[3][16384][512] | Qbf[3][512][512] | Wkt[3][...] | Wqt[3][...]
    //            (bf16) | u[8][512][16384] f32 | stats[8*512][2] f32
    __bf16* Kbf = (__bf16*)d_ws;
    __bf16* Qbf = Kbf + (size_t)3 * N_GENE * HD;
    __bf16* Wkt = Qbf + (size_t)3 * N_TF * HD;
    __bf16* Wqt = Wkt + (size_t)3 * HD * D_MODEL;
    float*  u_ws  = (float*)(Wqt + (size_t)3 * HD * D_MODEL);
    float*  stats = u_ws + (size_t)N_HEADS * N_TF * N_GENE;

    // pass-through outputs
    hipMemcpyAsync(out + OFF_HTF, H_TF, (size_t)CNT_HTF * sizeof(float),
                   hipMemcpyDeviceToDevice, stream);
    hipMemcpyAsync(out + OFF_HG, H_G, (size_t)CNT_HG * sizeof(float),
                   hipMemcpyDeviceToDevice, stream);

    // one-time transposed bf16 weights (L2-resident afterwards)
    for (int e = 0; e < 3; ++e) {
        wtrans_kernel<<<dim3(HD / 32, D_MODEL / 32), 256, 0, stream>>>(
            Wk[e], Wkt + (size_t)e * HD * D_MODEL);
        wtrans_kernel<<<dim3(HD / 32, D_MODEL / 32), 256, 0, stream>>>(
            Wq[e], Wqt + (size_t)e * HD * D_MODEL);
    }

    // projections: K_e = z_e @ Wk_e (bf16 out), Q_e = z_e[tf_idx] @ Wq_e (bf16 out)
    for (int e = 0; e < 3; ++e) {
        proj_gemm_kernel<<<dim3(HD / 128, N_GENE / 16), 256, 0, stream>>>(
            z[e], Wkt + (size_t)e * HD * D_MODEL, nullptr,
            Kbf + (size_t)e * N_GENE * HD);
        proj_gemm_kernel<<<dim3(HD / 128, N_TF / 16), 256, 0, stream>>>(
            z[e], Wqt + (size_t)e * HD * D_MODEL, tf_idx,
            Qbf + (size_t)e * N_TF * HD);
    }

    // fused all-head scores + gate (writes u[h], u_mean, alpha_mean once)
    score_gate_kernel<<<dim3(N_GENE / 128, N_TF / 16), 256, 0, stream>>>(
        Qbf, Kbf, gw, gb, u_ws, out + OFF_U, out + OFF_AL);

    // softmax over genes: per-row stats then single-store A_mean
    row_stats_kernel<<<N_HEADS * N_TF, 256, 0, stream>>>(u_ws, stats);
    amean_kernel<<<dim3(N_GENE / 256, N_TF), 256, 0, stream>>>(u_ws, stats,
                                                               out + OFF_A);
}
